// GraphLaplacianTransformerBackbone_57208964383552
// MI455X (gfx1250) — matrix operand order repaired and checked by reading, not compiled, not run on hardware
//
#include <hip/hip_runtime.h>

// ---------------- problem constants ----------------
#define NN   51200      // nodes
#define NE   409600     // edges
#define NB   64         // graphs / cls tokens
#define DIM  128
#define NH   8
#define HD   16
#define EDIM 32
#define NL   2
#define NLC  2
#define EHID 16
#define FFD  512
#define SCL  0.25f      // HD^-0.5

typedef unsigned short u16;
typedef unsigned int   u32;

typedef __attribute__((ext_vector_type(8)))  float  v8f;
typedef __attribute__((ext_vector_type(8)))  __bf16 v8bf;
typedef __attribute__((ext_vector_type(16))) __bf16 v16bf;

#if __has_builtin(__builtin_amdgcn_global_load_async_to_lds_b128) && \
    __has_builtin(__builtin_amdgcn_s_wait_asynccnt)
#define HAVE_ASYNC_LDS 1
// builtin signature (from hipcc diagnostic): param0 = v4i in global AS (non-const), param1 = LDS side
typedef int v4i_vec __attribute__((vector_size(16)));
typedef __attribute__((address_space(1))) v4i_vec g_v4i;   // global
typedef __attribute__((address_space(3))) v4i_vec l_v4i;   // LDS
#else
#define HAVE_ASYNC_LDS 0
#endif

// ---------------- device helpers ----------------
__device__ __forceinline__ u16 f2bf(float f) {
  u32 u = __float_as_uint(f);
  u += 0x7fffu + ((u >> 16) & 1u);       // round-to-nearest-even
  return (u16)(u >> 16);
}
__device__ __forceinline__ float gelu_t(float x) {
  float x3 = x * x * x;
  return 0.5f * x * (1.f + tanhf(0.7978845608028654f * (x + 0.044715f * x3)));
}
// monotonic float <-> uint map for atomicMax on floats
__device__ __forceinline__ u32 fenc(float f) {
  u32 b = __float_as_uint(f);
  return (b & 0x80000000u) ? ~b : (b | 0x80000000u);
}
__device__ __forceinline__ float fdec(u32 u) {
  u32 b = (u & 0x80000000u) ? (u & 0x7fffffffu) : ~u;
  return __uint_as_float(b);
}

// ---------------- WMMA GEMM ----------------
// C[M,Nc] = A[M,K](bf16) @ Wt[Nc,K](bf16, pre-transposed)
// Block: 128 thr (4 waves), tile 64(M) x 128(N); wave -> 16x128 (8 wmma tiles).
// B tile (128 rows x 32 K = 8KB) staged in LDS, double-buffered via async-to-LDS.
// EPI 0: Cf = scale * acc (f32)     EPI 1: Cb = bf16(gelu(acc + bias)) for FF1
union AFrag { v16bf v; v8bf h[2]; };

#define BSTRIDE 40   // u16 elements per LDS B row (80B: 16B-aligned, breaks 64-bank stride)

template<int EPI>
__global__ void __launch_bounds__(128)
gemm_wmma(const u16* __restrict__ A, const u16* __restrict__ Wt,
          const float* __restrict__ bias, float* __restrict__ Cf, u16* __restrict__ Cb,
          int M, int K, int Nc, float scale)
{
  __shared__ u16 Bs[2][128 * BSTRIDE];

  const int tid  = threadIdx.x;
  const int wv   = tid >> 5;
  const int lane = tid & 31;
  const int half = lane >> 4;
  const int l    = lane & 15;
  const int m0   = blockIdx.x * 64 + wv * 16;
  const int n0   = blockIdx.y * 128;

  v8f acc[8];
#pragma unroll
  for (int t = 0; t < 8; ++t)
#pragma unroll
    for (int r = 0; r < 8; ++r) acc[t][r] = 0.f;

  const u16* arow = A  + (size_t)(m0 + l) * K;
  const u16* bsrc = Wt + (size_t)(n0 + tid) * K;   // each thread stages one Wt row / K-step

  auto stage = [&](int buf, int kb) {
#if HAVE_ASYNC_LDS
    g_v4i* g = (g_v4i*)(bsrc + kb);
    l_v4i* s = (l_v4i*)(&Bs[buf][tid * BSTRIDE]);
    // INST_OFFSET applies to both global and LDS addresses (ISA 08_async_tensor §4.4)
    __builtin_amdgcn_global_load_async_to_lds_b128(g, s, 0,  0);
    __builtin_amdgcn_global_load_async_to_lds_b128(g, s, 16, 0);
    __builtin_amdgcn_global_load_async_to_lds_b128(g, s, 32, 0);
    __builtin_amdgcn_global_load_async_to_lds_b128(g, s, 48, 0);
#else
    const uint4* g = (const uint4*)(bsrc + kb);
    uint4* s = (uint4*)(&Bs[buf][tid * BSTRIDE]);
    s[0] = g[0]; s[1] = g[1]; s[2] = g[2]; s[3] = g[3];
#endif
  };

  const int nt = K >> 5;
  stage(0, 0);
  for (int i = 0; i < nt; ++i) {
    const int buf = i & 1;
    if (i + 1 < nt) {
      stage(buf ^ 1, (i + 1) << 5);
#if HAVE_ASYNC_LDS
      __builtin_amdgcn_s_wait_asynccnt(4);   // tile i's 4 loads done (in-order), tile i+1 in flight
#endif
    } else {
#if HAVE_ASYNC_LDS
      __builtin_amdgcn_s_wait_asynccnt(0);
#endif
    }
    __syncthreads();                          // tile i visible from all waves

    const int kb = i << 5;
    AFrag fa;
    // A layout (16-bit 16x32): halfwave 0 holds K 0..7 / 16..23, halfwave 1 holds K 8..15 / 24..31
    fa.h[0] = *(const v8bf*)(arow + kb + half * 8);
    fa.h[1] = *(const v8bf*)(arow + kb + 16 + half * 8);
    if (kb + 32 < K) __builtin_prefetch(arow + kb + 32, 0, 1);   // global_prefetch_b8

    const u16* bbase = &Bs[buf][0];
#pragma unroll
    for (int t = 0; t < 8; ++t) {
      // B layout (32x16): halfwave 0 holds K 0..7 / 16..23, halfwave 1 holds K 8..15 / 24..31
      AFrag fb;
      fb.h[0] = *(const v8bf*)(bbase + (t * 16 + l) * BSTRIDE + half * 8);
      fb.h[1] = *(const v8bf*)(bbase + (t * 16 + l) * BSTRIDE + 16 + half * 8);
      acc[t] = __builtin_amdgcn_wmma_f32_16x16x32_bf16(false, fa.v, false, fb.v, (short)0, acc[t], false, false);
    }
    __syncthreads();                          // safe to overwrite buf next iteration
  }

  const int row0 = m0 + half * 8;   // C layout: lane<16 -> M=r, lane>=16 -> M=8+r
#pragma unroll
  for (int t = 0; t < 8; ++t) {
    const int col = n0 + t * 16 + l;
    float bb = (EPI == 1) ? bias[col] : 0.f;
#pragma unroll
    for (int r = 0; r < 8; ++r) {
      float vv = acc[t][r] * scale;
      size_t idx = (size_t)(row0 + r) * Nc + col;
      if (EPI == 0) Cf[idx] = vv;
      else          Cb[idx] = f2bf(gelu_t(vv + bb));
    }
  }
}

// ---------------- small utility kernels ----------------
__global__ void fill_u32(u32* p, u32 v, long n) {
  long i = (long)blockIdx.x * blockDim.x + threadIdx.x;
  if (i < n) p[i] = v;
}

// f32 row-major W[K,Nc] -> bf16 transposed Wt[Nc,K]
__global__ void prep_wt(const float* __restrict__ W, u16* __restrict__ Wt, int K, int Nc) {
  int i = blockIdx.x * blockDim.x + threadIdx.x;
  if (i >= K * Nc) return;
  int k = i / Nc, n = i - k * Nc;
  Wt[(size_t)n * K + k] = f2bf(W[i]);
}

// LayerNorm rows -> bf16, one wave per row
__global__ void ln_bf16(const float* __restrict__ in, u16* __restrict__ out,
                        const float* __restrict__ w, const float* __restrict__ b,
                        int rows, int D)
{
  int wave = (int)(((long)blockIdx.x * blockDim.x + threadIdx.x) >> 5);
  if (wave >= rows) return;
  int lane = threadIdx.x & 31;
  const float* r = in + (size_t)wave * D;
  float s = 0.f, s2 = 0.f;
  for (int d = lane; d < D; d += 32) { float xv = r[d]; s += xv; s2 += xv * xv; }
#pragma unroll
  for (int o = 16; o; o >>= 1) { s += __shfl_xor(s, o, 32); s2 += __shfl_xor(s2, o, 32); }
  float invn = 1.f / (float)D;
  float mu  = s * invn;
  float var = s2 * invn - mu * mu;
  float inv = rsqrtf(var + 1e-5f);
  u16* orow = out + (size_t)wave * D;
  for (int d = lane; d < D; d += 32) orow[d] = f2bf((r[d] - mu) * inv * w[d] + b[d]);
}

// ---------------- edge attention passes (one wave per edge, coalesced row reads) ----------------
// pass A: logits per head, expand by Wexp (8->16), store araw, atomicMax per-src running max
__global__ void edge_pass_a(const float* __restrict__ q, const float* __restrict__ k,
                            const float* __restrict__ ek, const int* __restrict__ src,
                            const int* __restrict__ dst, const float* __restrict__ Wexp,
                            const float* __restrict__ temp, float* __restrict__ araw,
                            u32* __restrict__ mbits, int E)
{
  int e = (int)(((long)blockIdx.x * blockDim.x + threadIdx.x) >> 5);
  if (e >= E) return;
  int lane = threadIdx.x & 31;
  int s = src[e], d = dst[e];
  float4 q4 = ((const float4*)(q  + (size_t)s * DIM))[lane];
  float4 k4 = ((const float4*)(k  + (size_t)d * DIM))[lane];
  float4 e4 = ((const float4*)(ek + (size_t)e * DIM))[lane];
  float part = q4.x * (k4.x + e4.x) + q4.y * (k4.y + e4.y)
             + q4.z * (k4.z + e4.z) + q4.w * (k4.w + e4.w);
  part += __shfl_xor(part, 1, 32);   // reduce 4-lane head groups
  part += __shfl_xor(part, 2, 32);
  float logit = part * SCL / temp[0];          // head h = lane>>2, replicated over its 4 lanes
  float aj = 0.f;
  int j = lane & 15;
#pragma unroll
  for (int h = 0; h < NH; ++h)
    aj += __shfl(logit, h * 4, 32) * Wexp[h * EHID + j];
  if (lane < 16) {
    araw[(size_t)e * EHID + j] = aj;
    atomicMax(&mbits[(size_t)s * EHID + j], fenc(aj));
  }
}

// pass B: p = exp(araw - m[src]) (in-place), atomicAdd per-src sums
__global__ void edge_pass_b(const int* __restrict__ src, float* __restrict__ p,
                            const u32* __restrict__ mbits, float* __restrict__ ssum, int E)
{
  long t = (long)blockIdx.x * blockDim.x + threadIdx.x;
  int e = (int)(t >> 4);
  if (e >= E) return;
  int j = (int)(t & 15);
  int s = src[e];
  float m = fdec(mbits[(size_t)s * EHID + j]);
  float pv = __expf(p[(size_t)e * EHID + j] - m);
  p[(size_t)e * EHID + j] = pv;
  atomicAdd(&ssum[(size_t)s * EHID + j], pv);
}

// pass C: normalize, squeeze 16->8 via Wsq+bsq, scatter-add a*(v[dst]+ev) into agg[src]
__global__ void edge_pass_c(const int* __restrict__ src, const int* __restrict__ dst,
                            const float* __restrict__ p, const float* __restrict__ ssum,
                            const float* __restrict__ Wsq, const float* __restrict__ bsq,
                            const float* __restrict__ v, const float* __restrict__ ev,
                            float* __restrict__ agg, int E)
{
  int e = (int)(((long)blockIdx.x * blockDim.x + threadIdx.x) >> 5);
  if (e >= E) return;
  int lane = threadIdx.x & 31;
  int s = src[e], d = dst[e];
  float an = 0.f;
  if (lane < 16) an = p[(size_t)e * EHID + lane] / (ssum[(size_t)s * EHID + lane] + 1e-16f);
  int h = lane >> 2;
  float asq = bsq[h];
#pragma unroll
  for (int j = 0; j < EHID; ++j)
    asq += __shfl(an, j, 32) * Wsq[j * NH + h];
  float4 v4 = ((const float4*)(v  + (size_t)d * DIM))[lane];
  float4 e4 = ((const float4*)(ev + (size_t)e * DIM))[lane];
  float* ag = agg + (size_t)s * DIM + lane * 4;
  atomicAdd(ag + 0, asq * (v4.x + e4.x));
  atomicAdd(ag + 1, asq * (v4.y + e4.y));
  atomicAdd(ag + 2, asq * (v4.z + e4.z));
  atomicAdd(ag + 3, asq * (v4.w + e4.w));
}

// abuf = bf16(v - agg)  (GEMM input for Wo)
__global__ void vmagg_bf16(const float* __restrict__ v, const float* __restrict__ agg,
                           u16* __restrict__ out, long n)
{
  long i = (long)blockIdx.x * blockDim.x + threadIdx.x;
  if (i < n) out[i] = f2bf(v[i] - agg[i]);
}

// x += ls * (t + bias)
__global__ void residual(float* __restrict__ x, const float* __restrict__ t,
                         const float* __restrict__ ls, const float* __restrict__ bias, long n)
{
  long i = (long)blockIdx.x * blockDim.x + threadIdx.x;
  if (i >= n) return;
  int dd = (int)(i & (DIM - 1));
  x[i] += ls[dd] * (t[i] + bias[dd]);
}

// ---------------- cls pooling attention ----------------
__global__ void build_prefix(const int* __restrict__ portion, int* __restrict__ prefix) {
  if (blockIdx.x == 0 && threadIdx.x == 0) {
    int a = 0;
    for (int b = 0; b < NB; ++b) { prefix[b] = a; a += portion[b]; }
    prefix[NB] = a;
  }
}
__global__ void build_gid(const int* __restrict__ prefix, int* __restrict__ gid, int n) {
  int i = blockIdx.x * blockDim.x + threadIdx.x;
  if (i >= n) return;
  int lo = 0, hi = NB;
  while (hi - lo > 1) { int mid = (lo + hi) >> 1; if (prefix[mid] <= i) lo = mid; else hi = mid; }
  gid[i] = lo;
}
__global__ void cls_init(float* __restrict__ cls, const float* __restrict__ tok) {
  int i = blockIdx.x * blockDim.x + threadIdx.x;
  if (i < NB * DIM) cls[i] = tok[i & (DIM - 1)];
}
// e_c[b,h] = q[b,h].k_c[b,h]; seed running max
__global__ void cls_ec_init(const float* __restrict__ q, const float* __restrict__ k,
                            float* __restrict__ ec, u32* __restrict__ mb) {
  int i = blockIdx.x * blockDim.x + threadIdx.x;
  if (i >= NB * NH) return;
  int b = i >> 3, h = i & 7;
  const float* qr = q + (size_t)b * DIM + h * HD;
  const float* kr = k + (size_t)b * DIM + h * HD;
  float s = 0.f;
#pragma unroll
  for (int d = 0; d < HD; ++d) s += qr[d] * kr[d];
  ec[i] = s; mb[i] = fenc(s);
}
// e_n[i,h] = q[gid].k_n[i,h]; atomicMax into mb[gid]
__global__ void cls_en_max(const float* __restrict__ q, const float* __restrict__ k,
                           const int* __restrict__ gid, float* __restrict__ ens,
                           u32* __restrict__ mb, int n)
{
  int i = (int)(((long)blockIdx.x * blockDim.x + threadIdx.x) >> 5);
  if (i >= n) return;
  int lane = threadIdx.x & 31;
  int g = gid[i];
  float4 q4 = ((const float4*)(q + (size_t)g * DIM))[lane];
  float4 k4 = ((const float4*)(k + (size_t)(NB + i) * DIM))[lane];
  float part = q4.x * k4.x + q4.y * k4.y + q4.z * k4.z + q4.w * k4.w;
  part += __shfl_xor(part, 1, 32);
  part += __shfl_xor(part, 2, 32);
  if ((lane & 3) == 0) {
    int h = lane >> 2;
    ens[(size_t)i * NH + h] = part;
    atomicMax(&mb[g * NH + h], fenc(part));
  }
}
// p_n = exp(e_n - m[gid]); denom += p_n; num += p_n * v_n
__global__ void cls_pn_sum(const float* __restrict__ ens, const u32* __restrict__ mb,
                           const int* __restrict__ gid, const float* __restrict__ v,
                           float* __restrict__ denom, float* __restrict__ num, int n)
{
  int i = (int)(((long)blockIdx.x * blockDim.x + threadIdx.x) >> 5);
  if (i >= n) return;
  int lane = threadIdx.x & 31;
  int g = gid[i];
  int h = lane >> 2;
  float pn = __expf(ens[(size_t)i * NH + h] - fdec(mb[g * NH + h]));
  if ((lane & 3) == 0) atomicAdd(&denom[g * NH + h], pn);
  float4 v4 = ((const float4*)(v + (size_t)(NB + i) * DIM))[lane];
  float* nm = num + (size_t)g * DIM + lane * 4;
  atomicAdd(nm + 0, pn * v4.x);
  atomicAdd(nm + 1, pn * v4.y);
  atomicAdd(nm + 2, pn * v4.z);
  atomicAdd(nm + 3, pn * v4.w);
}
// pooled = (num + p_c*v_c) / (denom + p_c)  -> bf16 (GEMM input for Wo)
__global__ void cls_pool(const float* __restrict__ num, const float* __restrict__ denom,
                         const float* __restrict__ ec, const u32* __restrict__ mb,
                         const float* __restrict__ v, u16* __restrict__ pooled)
{
  int i = blockIdx.x * blockDim.x + threadIdx.x;
  if (i >= NB * DIM) return;
  int b = i >> 7, dd = i & (DIM - 1), h = dd >> 4;
  float m  = fdec(mb[b * NH + h]);
  float pc = __expf(ec[b * NH + h] - m);
  float nu = num[i] + pc * v[(size_t)b * DIM + dd];
  float de = denom[b * NH + h] + pc;
  pooled[i] = f2bf(nu / de);
}

// ---------------- host orchestration ----------------
static inline size_t alignup(size_t x) { return (x + 255) & ~(size_t)255; }

extern "C" void kernel_launch(void* const* d_in, const int* in_sizes, int n_in,
                              void* d_out, int out_size, void* d_ws, size_t ws_size,
                              hipStream_t stream)
{
  (void)in_sizes; (void)n_in; (void)out_size; (void)ws_size;
  // -------- inputs --------
  const float* x0    = (const float*)d_in[0];
  const float* edges = (const float*)d_in[1];
  const int*   src   = (const int*)d_in[2];
  const int*   dst   = src + NE;
  const int*   gport = (const int*)d_in[3];
  const float* clstk = (const float*)d_in[4];
  // token params (leading dim L=2)
  const float* t_lnxw = (const float*)d_in[5];
  const float* t_lnxb = (const float*)d_in[6];
  const float* t_lnew = (const float*)d_in[7];
  const float* t_lneb = (const float*)d_in[8];
  const float* t_Wq   = (const float*)d_in[9];
  const float* t_Wk   = (const float*)d_in[10];
  const float* t_Wv   = (const float*)d_in[11];
  const float* t_Wek  = (const float*)d_in[12];
  const float* t_Wev  = (const float*)d_in[13];
  const float* t_Wexp = (const float*)d_in[14];
  const float* t_Wsq  = (const float*)d_in[15];
  const float* t_bsq  = (const float*)d_in[16];
  const float* t_Wo   = (const float*)d_in[17];
  const float* t_bo   = (const float*)d_in[18];
  const float* t_temp = (const float*)d_in[19];
  const float* t_ls1  = (const float*)d_in[20];
  const float* t_flnw = (const float*)d_in[21];
  const float* t_flnb = (const float*)d_in[22];
  const float* t_W1   = (const float*)d_in[23];
  const float* t_b1   = (const float*)d_in[24];
  const float* t_W2   = (const float*)d_in[25];
  const float* t_b2   = (const float*)d_in[26];
  const float* t_ls2  = (const float*)d_in[27];
  // cls params (leading dim LC=2)
  const float* c_lnw  = (const float*)d_in[28];
  const float* c_lnb  = (const float*)d_in[29];
  const float* c_Wq   = (const float*)d_in[30];
  const float* c_Wk   = (const float*)d_in[31];
  const float* c_Wv   = (const float*)d_in[32];
  const float* c_Wo   = (const float*)d_in[33];
  const float* c_bo   = (const float*)d_in[34];
  const float* c_ls1  = (const float*)d_in[35];
  const float* c_flnw = (const float*)d_in[36];
  const float* c_flnb = (const float*)d_in[37];
  const float* c_W1   = (const float*)d_in[38];
  const float* c_b1   = (const float*)d_in[39];
  const float* c_W2   = (const float*)d_in[40];
  const float* c_b2   = (const float*)d_in[41];
  const float* c_ls2  = (const float*)d_in[42];

  // -------- workspace carve --------
  char* base = (char*)d_ws;
  size_t cur = 0;
  auto alloc = [&](size_t bytes) -> void* { void* p = base + cur; cur = alignup(cur + bytes); return p; };
  float* xcur = (float*)alloc((size_t)NN * DIM * 4);
  u16*   abuf = (u16*)  alloc((size_t)(NN + NB) * DIM * 2);   // bf16 activations (xn/hn/zn/vmagg/pooled)
  u16*   enbf = (u16*)  alloc((size_t)NE * EDIM * 2);
  float* qb   = (float*)alloc((size_t)(NN + NB) * DIM * 4);
  float* kb   = (float*)alloc((size_t)(NN + NB) * DIM * 4);
  float* vb   = (float*)alloc((size_t)(NN + NB) * DIM * 4);
  float* ekb  = (float*)alloc((size_t)NE * DIM * 4);
  float* evb  = (float*)alloc((size_t)NE * DIM * 4);
  float* pbuf = (float*)alloc((size_t)NE * EHID * 4);
  u32*   mbit = (u32*)  alloc((size_t)NN * EHID * 4);
  float* sbuf = (float*)alloc((size_t)NN * EHID * 4);
  float* agg  = (float*)alloc((size_t)NN * DIM * 4);
  u16*   gbuf = (u16*)  alloc((size_t)NN * FFD * 2);
  float* tbuf = (float*)alloc((size_t)NN * DIM * 4);
  float* ens  = (float*)alloc((size_t)NN * NH * 4);
  int*   pref = (int*)  alloc((NB + 1) * 4);
  int*   gid  = (int*)  alloc((size_t)NN * 4);
  float* ec   = (float*)alloc(NB * NH * 4);
  u32*   mbc  = (u32*)  alloc(NB * NH * 4);
  float* denc = (float*)alloc(NB * NH * 4);
  float* numc = (float*)alloc((size_t)NB * DIM * 4);
  u16* wtq  = (u16*)alloc(DIM * DIM * 2);
  u16* wtk  = (u16*)alloc(DIM * DIM * 2);
  u16* wtv  = (u16*)alloc(DIM * DIM * 2);
  u16* wto  = (u16*)alloc(DIM * DIM * 2);
  u16* wtek = (u16*)alloc(EDIM * DIM * 2);
  u16* wtev = (u16*)alloc(EDIM * DIM * 2);
  u16* wt1  = (u16*)alloc(DIM * FFD * 2);
  u16* wt2  = (u16*)alloc(FFD * DIM * 2);
  float* cls = (float*)d_out;                                  // running cls buffer IS the output

  // -------- launch helpers --------
  auto blocks = [](long n, int b) { return dim3((unsigned)((n + b - 1) / b)); };
  auto fill0  = [&](void* p, long n32) { fill_u32<<<blocks(n32, 256), 256, 0, stream>>>((u32*)p, 0u, n32); };
  auto prep   = [&](const float* W, u16* Wt, int K, int Nc) {
    prep_wt<<<blocks((long)K * Nc, 256), 256, 0, stream>>>(W, Wt, K, Nc);
  };
  auto ln = [&](const float* in, u16* out, const float* w, const float* b, int rows, int D) {
    ln_bf16<<<blocks((long)rows * 32, 256), 256, 0, stream>>>(in, out, w, b, rows, D);
  };
  auto gemm = [&](const u16* A, const u16* Wt, const float* bias, float* Cf, u16* Cb,
                  int M, int K, int Nc, float scale, int epi) {
    dim3 grid(M / 64, Nc / 128), blk(128);
    if (epi == 0) gemm_wmma<0><<<grid, blk, 0, stream>>>(A, Wt, bias, Cf, Cb, M, K, Nc, scale);
    else          gemm_wmma<1><<<grid, blk, 0, stream>>>(A, Wt, bias, Cf, Cb, M, K, Nc, scale);
  };

  // running node state
  (void)hipMemcpyAsync(xcur, x0, (size_t)NN * DIM * 4, hipMemcpyDeviceToDevice, stream);

  // ================= token layers =================
  for (int i = 0; i < NL; ++i) {
    prep(t_Wq  + (size_t)i * DIM * DIM,  wtq,  DIM,  DIM);
    prep(t_Wk  + (size_t)i * DIM * DIM,  wtk,  DIM,  DIM);
    prep(t_Wv  + (size_t)i * DIM * DIM,  wtv,  DIM,  DIM);
    prep(t_Wo  + (size_t)i * DIM * DIM,  wto,  DIM,  DIM);
    prep(t_Wek + (size_t)i * EDIM * DIM, wtek, EDIM, DIM);
    prep(t_Wev + (size_t)i * EDIM * DIM, wtev, EDIM, DIM);
    prep(t_W1  + (size_t)i * DIM * FFD,  wt1,  DIM,  FFD);
    prep(t_W2  + (size_t)i * FFD * DIM,  wt2,  FFD,  DIM);

    ln(xcur,  abuf, t_lnxw + i * DIM,  t_lnxb + i * DIM,  NN, DIM);
    ln(edges, enbf, t_lnew + i * EDIM, t_lneb + i * EDIM, NE, EDIM);

    gemm(abuf, wtq,  nullptr, qb,  nullptr, NN, DIM,  DIM, 1.f, 0);
    gemm(abuf, wtk,  nullptr, kb,  nullptr, NN, DIM,  DIM, 1.f, 0);
    gemm(abuf, wtv,  nullptr, vb,  nullptr, NN, DIM,  DIM, 1.f, 0);
    gemm(enbf, wtek, nullptr, ekb, nullptr, NE, EDIM, DIM, 1.f, 0);
    gemm(enbf, wtev, nullptr, evb, nullptr, NE, EDIM, DIM, 1.f, 0);

    fill0(mbit, (long)NN * EHID);
    fill0(sbuf, (long)NN * EHID);
    fill0(agg,  (long)NN * DIM);

    edge_pass_a<<<blocks((long)NE * 32, 256), 256, 0, stream>>>(
        qb, kb, ekb, src, dst, t_Wexp + (size_t)i * NH * EHID, t_temp + i, pbuf, mbit, NE);
    edge_pass_b<<<blocks((long)NE * EHID, 256), 256, 0, stream>>>(src, pbuf, mbit, sbuf, NE);
    edge_pass_c<<<blocks((long)NE * 32, 256), 256, 0, stream>>>(
        src, dst, pbuf, sbuf, t_Wsq + (size_t)i * EHID * NH, t_bsq + i * NH, vb, evb, agg, NE);

    vmagg_bf16<<<blocks((long)NN * DIM, 256), 256, 0, stream>>>(vb, agg, abuf, (long)NN * DIM);
    gemm(abuf, wto, nullptr, tbuf, nullptr, NN, DIM, DIM, 1.f, 0);
    residual<<<blocks((long)NN * DIM, 256), 256, 0, stream>>>(
        xcur, tbuf, t_ls1 + i * DIM, t_bo + i * DIM, (long)NN * DIM);

    ln(xcur, abuf, t_flnw + i * DIM, t_flnb + i * DIM, NN, DIM);
    gemm(abuf, wt1, t_b1 + (size_t)i * FFD, nullptr, gbuf, NN, DIM, FFD, 1.f, 1);
    gemm(gbuf, wt2, nullptr, tbuf, nullptr, NN, FFD, DIM, 1.f, 0);
    residual<<<blocks((long)NN * DIM, 256), 256, 0, stream>>>(
        xcur, tbuf, t_ls2 + i * DIM, t_b2 + i * DIM, (long)NN * DIM);
  }

  // ================= cls pooling layers =================
  build_prefix<<<1, 1, 0, stream>>>(gport, pref);
  build_gid<<<blocks(NN, 256), 256, 0, stream>>>(pref, gid, NN);
  cls_init<<<blocks(NB * DIM, 256), 256, 0, stream>>>(cls, clstk);

  for (int i = 0; i < NLC; ++i) {
    prep(c_Wq + (size_t)i * DIM * DIM, wtq, DIM, DIM);
    prep(c_Wk + (size_t)i * DIM * DIM, wtk, DIM, DIM);
    prep(c_Wv + (size_t)i * DIM * DIM, wtv, DIM, DIM);
    prep(c_Wo + (size_t)i * DIM * DIM, wto, DIM, DIM);
    prep(c_W1 + (size_t)i * DIM * FFD, wt1, DIM, FFD);
    prep(c_W2 + (size_t)i * FFD * DIM, wt2, FFD, DIM);

    // zn = LN(concat(cls, x)) -> abuf rows [0,NB) and [NB, NB+NN)
    ln(cls,  abuf,                    c_lnw + i * DIM, c_lnb + i * DIM, NB, DIM);
    ln(xcur, abuf + (size_t)NB * DIM, c_lnw + i * DIM, c_lnb + i * DIM, NN, DIM);

    gemm(abuf, wtq, nullptr, qb, nullptr, NB,      DIM, DIM, 1.f, 0);   // q: cls rows only
    gemm(abuf, wtk, nullptr, kb, nullptr, NN + NB, DIM, DIM, SCL, 0);   // k pre-scaled
    gemm(abuf, wtv, nullptr, vb, nullptr, NN + NB, DIM, DIM, 1.f, 0);

    cls_ec_init<<<blocks(NB * NH, 256), 256, 0, stream>>>(qb, kb, ec, mbc);
    fill0(denc, NB * NH);
    fill0(numc, (long)NB * DIM);
    cls_en_max<<<blocks((long)NN * 32, 256), 256, 0, stream>>>(qb, kb, gid, ens, mbc, NN);
    cls_pn_sum<<<blocks((long)NN * 32, 256), 256, 0, stream>>>(ens, mbc, gid, vb, denc, numc, NN);
    cls_pool<<<blocks(NB * DIM, 256), 256, 0, stream>>>(numc, denc, ec, mbc, vb, abuf);

    gemm(abuf, wto, nullptr, tbuf, nullptr, NB, DIM, DIM, 1.f, 0);
    residual<<<blocks((long)NB * DIM, 256), 256, 0, stream>>>(
        cls, tbuf, c_ls1 + i * DIM, c_bo + i * DIM, (long)NB * DIM);

    ln(cls, abuf, c_flnw + i * DIM, c_flnb + i * DIM, NB, DIM);
    gemm(abuf, wt1, c_b1 + (size_t)i * FFD, nullptr, gbuf, NB, DIM, FFD, 1.f, 1);
    gemm(gbuf, wt2, nullptr, tbuf, nullptr, NB, FFD, DIM, 1.f, 0);
    residual<<<blocks((long)NB * DIM, 256), 256, 0, stream>>>(
        cls, tbuf, c_ls2 + i * DIM, c_b2 + i * DIM, (long)NB * DIM);
  }
}
